// TensorProductDecoder_28501402977088
// MI455X (gfx1250) — matrix operand order repaired and checked by reading, not compiled, not run on hardware
//
#include <hip/hip_runtime.h>
#include <hip/hip_bf16.h>
#include <math.h>

typedef __bf16 bf16_t;
typedef bf16_t v16bf __attribute__((ext_vector_type(16)));
typedef float  v8f   __attribute__((ext_vector_type(8)));

#define DEV static __device__ __forceinline__

// ---------- bf16 helpers: use native hardware converts (v_cvt_*bf16*) ----------
DEV unsigned short f2bf_u(float f) {
  bf16_t b = (bf16_t)f;                       // hardware cvt, RNE
  union { bf16_t b; unsigned short s; } u; u.b = b;
  return u.s;
}
DEV float bf2f(unsigned short s) {
  union { unsigned short s; bf16_t b; } u; u.s = s;
  return (float)u.b;
}

// ---------- WMMA fragment loads ----------
// B fragment (32x16 K x N, bf16): lane serves column n = lane&15; half h = lane>>4
// holds K = [k0 + 16h, k0 + 16h + 16) -> one contiguous 32B region.
DEV v16bf loadB_bf(const unsigned short* colrow, int k0, int h) {
  v16bf r;
  uint4* q = reinterpret_cast<uint4*>(&r);
  const unsigned short* p = colrow + k0 + 16 * h;
  q[0] = *reinterpret_cast<const uint4*>(p);
  q[1] = *reinterpret_cast<const uint4*>(p + 8);
  return r;
}
// A fragment (16x32 M x K, bf16): lane row m = lane&15; half h = lane>>4;
// elements 0..15 = K [k0+16h, +16).
DEV v16bf loadA_bf(const unsigned short* row, int k0, int h) {
  v16bf r;
  uint4* q = reinterpret_cast<uint4*>(&r);
  q[0] = *reinterpret_cast<const uint4*>(row + k0 + 16 * h);
  q[1] = *reinterpret_cast<const uint4*>(row + k0 + 16 * h + 8);
  return r;
}

DEV v8f wmma_bf16(v16bf a, v16bf b, v8f c) {
  return __builtin_amdgcn_wmma_f32_16x16x32_bf16(false, a, false, b, (short)0, c, false, false);
}

// ---------- prep kernels ----------
__global__ void cvt_f32_bf16(const float* __restrict__ src,
                             unsigned short* __restrict__ dst, int n, int zero_n) {
  int i = blockIdx.x * blockDim.x + threadIdx.x;
  int stride = gridDim.x * blockDim.x;
  for (; i < n; i += stride) {
    float v = (i < zero_n) ? 0.f : src[i];
    dst[i] = f2bf_u(v);
  }
}

__global__ void rownorm_f32(const float* __restrict__ src, float* __restrict__ norms,
                            int nrows, int ncols) {
  int wave = threadIdx.x >> 5, lane = threadIdx.x & 31;
  int row = blockIdx.x * 8 + wave;
  if (row >= nrows) return;
  const float* r = src + (size_t)row * ncols;
  float s = 0.f;
  for (int i = lane; i < ncols; i += 32) { float v = r[i]; s += v * v; }
  for (int off = 16; off; off >>= 1) s += __shfl_down(s, off, 32);
  if (lane == 0) norms[row] = s;
}

__global__ void rownorm_bf16(const unsigned short* __restrict__ src, float* __restrict__ norms,
                             int nrows, int ncols) {
  int wave = threadIdx.x >> 5, lane = threadIdx.x & 31;
  int row = blockIdx.x * 8 + wave;
  if (row >= nrows) return;
  const unsigned short* r = src + (size_t)row * ncols;
  float s = 0.f;
  for (int i = lane; i < ncols; i += 32) { float v = bf2f(r[i]); s += v * v; }
  for (int off = 16; off; off >>= 1) s += __shfl_down(s, off, 32);
  if (lane == 0) norms[row] = s;
}

// ---------- GEMM1: H(64x65536) = hidden(64x1024) @ W^T + b, bf16 out ----------
// Bandwidth-bound on streaming W (256 MB). B column n == W row n (contiguous);
// W converted f32->bf16 in-register with native cvt.
__global__ __launch_bounds__(256) void gemm1_hWb(
    const unsigned short* __restrict__ hid,   // 64x1024 bf16
    const float* __restrict__ W,              // 65536x1024 f32
    const float* __restrict__ bias,           // 65536 f32
    unsigned short* __restrict__ H)           // 64x65536 bf16 == (b, f, r)
{
  const int wave = threadIdx.x >> 5, lane = threadIdx.x & 31;
  const int ml = lane & 15, h = lane >> 4;
  const int n  = blockIdx.x * 128 + wave * 16 + ml;
  const int m0 = blockIdx.y * 16;
  const unsigned short* arow = hid + (size_t)(m0 + ml) * 1024;
  const float* brow = W + (size_t)n * 1024;
  v8f acc = {};
  #pragma unroll 2
  for (int k0 = 0; k0 < 1024; k0 += 32) {
    v16bf a = loadA_bf(arow, k0, h);
    // B lane: 16 contiguous f32 of W row n at k0+16h, native cvt to bf16
    const float4* bp = reinterpret_cast<const float4*>(brow + k0 + 16 * h);
    float4 f0 = bp[0], f1 = bp[1], f2 = bp[2], f3 = bp[3];
    float fv[16] = {f0.x,f0.y,f0.z,f0.w, f1.x,f1.y,f1.z,f1.w,
                    f2.x,f2.y,f2.z,f2.w, f3.x,f3.y,f3.z,f3.w};
    v16bf b;
    bf16_t* bs = reinterpret_cast<bf16_t*>(&b);
    #pragma unroll
    for (int i = 0; i < 16; ++i) bs[i] = (bf16_t)fv[i];
    acc = wmma_bf16(a, b, acc);
  }
  const float bv = bias[n];
  #pragma unroll
  for (int j = 0; j < 8; ++j) {                 // C: lane holds (m = m0+j+8h, n)
    int m = m0 + j + 8 * h;
    H[(size_t)m * 65536 + n] = f2bf_u(acc[j] + bv);
  }
}

// ---------- GEMM2: fg[b](128x256) = roleE[roles[b,:]](128x256) @ H[b]^T ----------
// B column n=f is H[b, f, :] (contiguous row of H). A rows gathered by role index.
__global__ __launch_bounds__(256) void gemm2_fg(
    const int* __restrict__ roles,            // 64x128 int32
    const unsigned short* __restrict__ roleE, // 256x256 bf16 (row0 zeroed)
    const unsigned short* __restrict__ H,     // 64x256x256 bf16
    unsigned short* __restrict__ fg)          // 8192x256 bf16
{
  const int wave = threadIdx.x >> 5, lane = threadIdx.x & 31;
  const int ml = lane & 15, h = lane >> 4;
  const int b  = blockIdx.z;
  const int n  = blockIdx.x * 128 + wave * 16 + ml;  // f
  const int s  = blockIdx.y * 16 + ml;               // A row within batch
  const int ridx = roles[b * 128 + s];
  const unsigned short* arow = roleE + (size_t)ridx * 256;
  const unsigned short* brow = H + (size_t)b * 65536 + (size_t)n * 256;
  v8f acc = {};
  #pragma unroll
  for (int k0 = 0; k0 < 256; k0 += 32) {
    v16bf a  = loadA_bf(arow, k0, h);
    v16bf bm = loadB_bf(brow, k0, h);
    acc = wmma_bf16(a, bm, acc);
  }
  #pragma unroll
  for (int j = 0; j < 8; ++j) {
    int m = blockIdx.y * 16 + j + 8 * h;
    fg[(size_t)(b * 128 + m) * 256 + n] = f2bf_u(acc[j]);
  }
}

// ---------- GEMM3: -dist(8192x10000), 2x N register blocking ----------
// Each wave computes a 16x32 tile: A fragment reused across two B fragments.
// N padded to 10240; out-of-range columns clamp to filler row 0 so EXEC stays
// all-ones for WMMA; their stores are skipped.
__global__ __launch_bounds__(256) void gemm3_dist(
    const unsigned short* __restrict__ fg,    // 8192x256 bf16
    const unsigned short* __restrict__ fil,   // 10000x256 bf16
    const float* __restrict__ fgn,            // 8192 row norms^2
    const float* __restrict__ fn,             // 10000 filler norms^2
    float* __restrict__ out)                  // 8192x10000, stores -dist
{
  const int wave = threadIdx.x >> 5, lane = threadIdx.x & 31;
  const int ml = lane & 15, h = lane >> 4;
  const int nbase = blockIdx.x * 256 + wave * 32;
  const int n0 = nbase + ml, n1 = nbase + 16 + ml;
  const bool valid0 = (n0 < 10000), valid1 = (n1 < 10000);
  const int bn0 = valid0 ? n0 : 0, bn1 = valid1 ? n1 : 0;
  const int m0 = blockIdx.y * 16;
  const unsigned short* arow  = fg  + (size_t)(m0 + ml) * 256;
  const unsigned short* brow0 = fil + (size_t)bn0 * 256;
  const unsigned short* brow1 = fil + (size_t)bn1 * 256;
  v8f acc0 = {}, acc1 = {};
  #pragma unroll
  for (int k0 = 0; k0 < 256; k0 += 32) {
    v16bf a = loadA_bf(arow, k0, h);
    acc0 = wmma_bf16(a, loadB_bf(brow0, k0, h), acc0);
    acc1 = wmma_bf16(a, loadB_bf(brow1, k0, h), acc1);
  }
  const float fnv0 = fn[bn0], fnv1 = fn[bn1];
  #pragma unroll
  for (int j = 0; j < 8; ++j) {
    const int m = m0 + j + 8 * h;
    const float g = fgn[m];
    if (valid0) {
      float d2 = g - 2.f * acc0[j] + fnv0;
      out[(size_t)m * 10000 + n0] = -sqrtf(fmaxf(d2, 0.f));
    }
    if (valid1) {
      float d2 = g - 2.f * acc1[j] + fnv1;
      out[(size_t)m * 10000 + n1] = -sqrtf(fmaxf(d2, 0.f));
    }
  }
}

// ---------- per-row log_softmax + argmax; row cached in LDS (40KB << 320KB) ----------
__global__ __launch_bounds__(256) void logsoftmax_rows(float* __restrict__ out,
                                                       float* __restrict__ preds, int N) {
  __shared__ float srow[10000];
  __shared__ float rv[256];
  __shared__ int   ri[256];
  const int t = threadIdx.x;
  float* row = out + (size_t)blockIdx.x * N;
  float lmax = -3.4e38f; int larg = 0;
  for (int i = t; i < N; i += 256) {
    float v = row[i]; srow[i] = v;
    if (v > lmax) { lmax = v; larg = i; }
  }
  rv[t] = lmax; ri[t] = larg; __syncthreads();
  for (int s = 128; s > 0; s >>= 1) {
    if (t < s) {
      float ov = rv[t + s]; int oi = ri[t + s];
      if (ov > rv[t] || (ov == rv[t] && oi < ri[t])) { rv[t] = ov; ri[t] = oi; }
    }
    __syncthreads();
  }
  const float m = rv[0]; const int arg = ri[0];
  __syncthreads();
  float ls = 0.f;
  for (int i = t; i < N; i += 256) ls += expf(srow[i] - m);
  rv[t] = ls; __syncthreads();
  for (int s = 128; s > 0; s >>= 1) { if (t < s) rv[t] += rv[t + s]; __syncthreads(); }
  const float lz = m + logf(rv[0]);     // logsumexp
  for (int i = t; i < N; i += 256) row[i] = srow[i] - lz;
  if (t == 0) preds[blockIdx.x] = (float)arg;
}

// ---------- host ----------
extern "C" void kernel_launch(void* const* d_in, const int* in_sizes, int n_in,
                              void* d_out, int out_size, void* d_ws, size_t ws_size,
                              hipStream_t stream) {
  const int*   roles  = (const int*)  d_in[0];  // (64,128) int32
  const float* hidden = (const float*)d_in[1];  // (1,64,1024)
  const float* filler = (const float*)d_in[2];  // (10000,256)
  const float* roleE  = (const float*)d_in[3];  // (256,256)
  const float* W      = (const float*)d_in[4];  // (65536,1024)
  const float* bias   = (const float*)d_in[5];  // (65536,)
  float* out = (float*)d_out;

  char* ws = (char*)d_ws;
  size_t off = 0;
  auto alloc = [&](size_t bytes) -> char* {
    char* p = ws + off;
    off = (off + bytes + 255) & ~(size_t)255;
    return p;
  };
  unsigned short* role_bf = (unsigned short*)alloc(65536ull * 2);
  unsigned short* hid_bf  = (unsigned short*)alloc(65536ull * 2);
  unsigned short* fil_bf  = (unsigned short*)alloc(2560000ull * 2);
  float*          fil_n   = (float*)         alloc(10000ull * 4);
  unsigned short* H_bf    = (unsigned short*)alloc(4194304ull * 2);   // 64*65536
  unsigned short* fg_bf   = (unsigned short*)alloc(2097152ull * 2);   // 8192*256
  float*          fg_n    = (float*)         alloc(8192ull * 4);

  cvt_f32_bf16<<<256, 256, 0, stream>>>(roleE,  role_bf, 65536, 256);  // zero row 0 (padding_idx)
  cvt_f32_bf16<<<256, 256, 0, stream>>>(hidden, hid_bf,  65536, 0);
  cvt_f32_bf16<<<2048, 256, 0, stream>>>(filler, fil_bf, 2560000, 0);
  rownorm_f32<<<1250, 256, 0, stream>>>(filler, fil_n, 10000, 256);

  gemm1_hWb<<<dim3(512, 4), 256, 0, stream>>>(hid_bf, W, bias, H_bf);
  gemm2_fg<<<dim3(2, 8, 64), 256, 0, stream>>>(roles, role_bf, H_bf, fg_bf);
  rownorm_bf16<<<1024, 256, 0, stream>>>(fg_bf, fg_n, 8192, 256);
  gemm3_dist<<<dim3(40, 512), 256, 0, stream>>>(fg_bf, fil_bf, fg_n, fil_n, out);
  logsoftmax_rows<<<8192, 256, 0, stream>>>(out, out + 81920000ull, 10000);

  (void)in_sizes; (void)n_in; (void)out_size; (void)ws_size;
}
